// AdditiveAttention_72679436582936
// MI455X (gfx1250) — compile-verified, matrix operand order checked by reading, not run on hardware
//
#include <hip/hip_runtime.h>

#define B_ 64
#define T_ 4096
#define D_ENC_ 512
#define D_ATT_ 256
#define TILE_T 128
#define NT (T_ / TILE_T)   // 32 tiles per batch

typedef __attribute__((ext_vector_type(16))) _Float16 v16h;
typedef __attribute__((ext_vector_type(8)))  _Float16 v8h;
typedef __attribute__((ext_vector_type(8)))  float    v8f;
typedef __attribute__((ext_vector_type(4)))  float    v4f;

#if __has_builtin(__builtin_amdgcn_tanhf)
#define TANHF(x) __builtin_amdgcn_tanhf(x)
#else
#define TANHF(x) tanhf(x)
#endif

// ws layout (bytes)
#define WS_BPACK   0            // 256 KB  f16 Wk in B-fragment order
#define WS_QFOLD   262144       // 64 KB   fp32 qfold[B][256]
#define WS_SCORES  327680       // 1 MB    fp32 raw masked scores[B][T]
#define WS_CTXP    1376256      // 4 MB    fp32 ctxp[B][NT][512]
#define WS_MZ      5570560      // 16 KB   fp32 (m_tile, Z_tile)[B][NT][2]

// dynamic LDS layout (bytes) for scores_ctx_kernel.
// NOTE: kernel has NO static __shared__, so the dynamic segment begins at LDS
// offset 0 and these constants double as raw LDS addresses for async-to-LDS.
#define SM_TILE    0            // 128*512*2 = 131072 : f16 h_enc tile
#define SM_S       131072       // 128*4            : masked scores
#define SM_P       131584       // 128*4            : exp weights
#define SM_BC      132096       // 16 B             : broadcast slots
#define SM_QF      132112       // 1024 B           : qfold row (async-loaded)
#define SM_VW      133136       // 1024 B           : v_w      (async-loaded)
#define SM_BYTES   134160

// ---------------------------------------------------------------------------
// K0: repack Wk fp32 [256,512] -> f16 in WMMA B-fragment order (lane-contig).
// ---------------------------------------------------------------------------
__global__ __launch_bounds__(256) void pack_wk(const float* __restrict__ Wk,
                                               _Float16* __restrict__ Bpack) {
    int idx = blockIdx.x * blockDim.x + threadIdx.x;   // 0 .. 131071
    int i    =  idx        & 15;
    int lane = (idx >> 4)  & 31;
    int kt   = (idx >> 9)  & 15;
    int n    = (idx >> 13) & 15;
    int kb   = (lane < 16) ? 0 : 8;
    int koff = (i < 8) ? (kb + i) : (kb + 16 + (i - 8));
    int k = kt * 32 + koff;             // 0..511  (encoder dim)
    int a = n * 16 + (lane & 15);       // 0..255  (attention dim)
    Bpack[idx] = (_Float16)Wk[a * D_ENC_ + k];
}

// ---------------------------------------------------------------------------
// K1: qfold[b][a] = sum_e Wq[a][e]*h_dec[b][e] + Wq_b[a] + Wk_b[a]
// ---------------------------------------------------------------------------
__global__ __launch_bounds__(256) void qfold_kernel(const float* __restrict__ h_dec,
                                                    const float* __restrict__ Wq,
                                                    const float* __restrict__ Wq_b,
                                                    const float* __restrict__ Wk_b,
                                                    float* __restrict__ qfold) {
    int b = blockIdx.x;
    int a = threadIdx.x;
    float acc = Wq_b[a] + Wk_b[a];
    const float* hd = h_dec + b * D_ENC_;
    const float* wr = Wq + a * D_ENC_;
    for (int e = 0; e < D_ENC_; ++e) acc += wr[e] * hd[e];
    qfold[b * D_ATT_ + a] = acc;
}

// ---------------------------------------------------------------------------
// K2 (fused): per (batch, 128-row tile):
//   - wave 0 async-copies qfold row + v_w into LDS (overlaps the WMMA loop)
//   - stream h_enc rows fp32->f16 into WMMA A-frags AND into an LDS f16 tile
//   - k = tile @ Wk^T via 256x v_wmma_f32_16x16x32_f16
//   - s = v . tanh(k + qfold), masked; tile-local softmax (m_tile, Z_tile)
//   - unnormalized partial ctx from the LDS tile  (single pass over h_enc)
// ---------------------------------------------------------------------------
__global__ __launch_bounds__(256) void scores_ctx_kernel(
    const float* __restrict__ h_enc, const _Float16* __restrict__ Bpack,
    const float* __restrict__ qfold, const float* __restrict__ v_w,
    const unsigned char* __restrict__ mask, float* __restrict__ scores,
    float* __restrict__ ctxp, float* __restrict__ mz) {
    extern __shared__ char smem[];
    _Float16* tile  = (_Float16*)(smem + SM_TILE);   // [128][512]
    float*    s_lds = (float*)(smem + SM_S);         // [128]
    float*    p_lds = (float*)(smem + SM_P);         // [128]
    float*    bc    = (float*)(smem + SM_BC);
    float*    qf_l  = (float*)(smem + SM_QF);        // [256]
    float*    vw_l  = (float*)(smem + SM_VW);        // [256]

    int bid = blockIdx.x;
    int b   = bid / NT;
    int tileIdx = bid % NT;
    int t0  = tileIdx * TILE_T;
    int tid = threadIdx.x;
    int w    = tid >> 5;
    int l    = tid & 31;
    int l15  = l & 15;
    int half = l >> 4;
    int kb   = half ? 8 : 0;

    // Wave 0: async-stage qfold[b][0..255] and v_w[0..255] into LDS.
    // Per lane: 16B; offset: applies to both the LDS and global address.
    if (w == 0) {
        unsigned qofs = SM_QF + (unsigned)l * 16u;
        unsigned long long qga =
            (unsigned long long)(uintptr_t)(qfold + b * D_ATT_) + (unsigned long long)l * 16ull;
        asm volatile("global_load_async_to_lds_b128 %0, %1, off"
                     :: "v"(qofs), "v"(qga) : "memory");
        asm volatile("global_load_async_to_lds_b128 %0, %1, off offset:512"
                     :: "v"(qofs), "v"(qga) : "memory");
        unsigned vofs = SM_VW + (unsigned)l * 16u;
        unsigned long long vga =
            (unsigned long long)(uintptr_t)(v_w) + (unsigned long long)l * 16ull;
        asm volatile("global_load_async_to_lds_b128 %0, %1, off"
                     :: "v"(vofs), "v"(vga) : "memory");
        asm volatile("global_load_async_to_lds_b128 %0, %1, off offset:512"
                     :: "v"(vofs), "v"(vga) : "memory");
    }

    int row_l = 16 * w + l15;                       // local tile row 0..127
    const float* rowp = h_enc + ((size_t)b * T_ + (t0 + row_l)) * (size_t)D_ENC_;

    v8f acc[16] = {};

#pragma unroll
    for (int kt = 0; kt < 16; ++kt) {
        const float* p = rowp + kt * 32 + kb;
        if (kt < 15) __builtin_prefetch(rowp + (kt + 1) * 32 + kb, 0, 0);
        v4f f0 = *(const v4f*)(p);
        v4f f1 = *(const v4f*)(p + 4);
        v4f f2 = *(const v4f*)(p + 16);
        v4f f3 = *(const v4f*)(p + 20);
        v8h lo, hi;
#pragma unroll
        for (int i = 0; i < 4; ++i) {
            lo[i]     = (_Float16)f0[i];
            lo[4 + i] = (_Float16)f1[i];
            hi[i]     = (_Float16)f2[i];
            hi[4 + i] = (_Float16)f3[i];
        }
        // collaboratively retain the f16 tile in LDS (each element written once)
        *(v8h*)(tile + (size_t)row_l * D_ENC_ + kt * 32 + kb)      = lo;
        *(v8h*)(tile + (size_t)row_l * D_ENC_ + kt * 32 + kb + 16) = hi;
        v16h afrag = __builtin_shufflevector(lo, hi, 0, 1, 2, 3, 4, 5, 6, 7,
                                             8, 9, 10, 11, 12, 13, 14, 15);
#pragma unroll
        for (int n = 0; n < 16; ++n) {
            v16h bfrag = *(const v16h*)(Bpack + (((n * 16 + kt) * 32 + l) << 4));
            acc[n] = __builtin_amdgcn_wmma_f32_16x16x32_f16(
                false, afrag, false, bfrag, (short)0, acc[n], false, false);
        }
    }

    // make wave 0's async staging visible to the whole block
    if (w == 0) asm volatile("s_wait_asynccnt 0x0" ::: "memory");
    __syncthreads();

    // Epilogue: s[t] = sum_a tanh(k[t][a] + q[a]) * v[a]   (q, v from LDS)
    float sacc[8];
#pragma unroll
    for (int r = 0; r < 8; ++r) sacc[r] = 0.0f;
#pragma unroll
    for (int n = 0; n < 16; ++n) {
        int a_idx = n * 16 + l15;
        float qv = qf_l[a_idx];
        float vw = vw_l[a_idx];
#pragma unroll
        for (int r = 0; r < 8; ++r)
            sacc[r] += TANHF(acc[n][r] + qv) * vw;
    }
#pragma unroll
    for (int r = 0; r < 8; ++r) {
        float v = sacc[r];
        v += __shfl_xor(v, 1, 32);
        v += __shfl_xor(v, 2, 32);
        v += __shfl_xor(v, 4, 32);
        v += __shfl_xor(v, 8, 32);   // reduced within each 16-lane half
        if (l15 == 0) {
            int t_loc = 16 * w + half * 8 + r;
            int t = t0 + t_loc;
            float s = mask[(size_t)b * T_ + t] ? v : -1e9f;
            scores[(size_t)b * T_ + t] = s;   // raw masked score for final w
            s_lds[t_loc] = s;
        }
    }
    __syncthreads();

    // tile-local max (wave 0)
    if (tid < 32) {
        float m = -3.0e38f;
#pragma unroll
        for (int i = 0; i < 4; ++i) m = fmaxf(m, s_lds[tid * 4 + i]);
        for (int mk = 16; mk >= 1; mk >>= 1) m = fmaxf(m, __shfl_xor(m, mk, 32));
        if (tid == 0) bc[0] = m;
    }
    __syncthreads();
    float m_tile = bc[0];
    if (tid < 128) p_lds[tid] = expf(s_lds[tid] - m_tile);
    __syncthreads();
    if (tid < 32) {
        float z = 0.0f;
#pragma unroll
        for (int i = 0; i < 4; ++i) z += p_lds[tid * 4 + i];
        for (int mk = 16; mk >= 1; mk >>= 1) z += __shfl_xor(z, mk, 32);
        if (tid == 0) {
            mz[(size_t)(b * NT + tileIdx) * 2 + 0] = m_tile;
            mz[(size_t)(b * NT + tileIdx) * 2 + 1] = z;
        }
    }
    __syncthreads();

    // unnormalized partial ctx: ctxp[e] = sum_t p[t] * h16[t][e]
    int e2 = tid * 2;                 // two consecutive f16 per thread
    float a0 = 0.0f, a1 = 0.0f;
    for (int t = 0; t < TILE_T; ++t) {
        float pv = p_lds[t];
        unsigned int hv = *(const unsigned int*)(tile + (size_t)t * D_ENC_ + e2);
        _Float16 h0 = __builtin_bit_cast(_Float16, (unsigned short)(hv & 0xffffu));
        _Float16 h1 = __builtin_bit_cast(_Float16, (unsigned short)(hv >> 16));
        a0 += pv * (float)h0;
        a1 += pv * (float)h1;
    }
    float* cp = ctxp + (size_t)(b * NT + tileIdx) * D_ENC_;
    cp[e2]     = a0;
    cp[e2 + 1] = a1;
}

// ---------------------------------------------------------------------------
// K3: per-batch finalize: merge tile partials (flash-attention rescale),
// write ctx and normalized w.
// ---------------------------------------------------------------------------
__global__ __launch_bounds__(256) void finalize_kernel(
    const float* __restrict__ scores, const float* __restrict__ ctxp,
    const float* __restrict__ mz, float* __restrict__ ctx_out,
    float* __restrict__ w_out) {
    int b = blockIdx.x;
    int tid = threadIdx.x;
    __shared__ float scale[NT];
    __shared__ float bc[2];

    if (tid < 32) {
        float mi = mz[(size_t)(b * NT + tid) * 2 + 0];
        float zi = mz[(size_t)(b * NT + tid) * 2 + 1];
        float m = mi;
        for (int mk = 16; mk >= 1; mk >>= 1) m = fmaxf(m, __shfl_xor(m, mk, 32));
        float sc = expf(mi - m);
        scale[tid] = sc;
        float z = zi * sc;
        for (int mk = 16; mk >= 1; mk >>= 1) z += __shfl_xor(z, mk, 32);
        if (tid == 0) { bc[0] = m; bc[1] = z; }
    }
    __syncthreads();
    float m = bc[0];
    float rz = 1.0f / bc[1];

    // ctx[b][e] = (1/Z) * sum_tiles scale[i] * ctxp[b][i][e]
    for (int e = tid; e < D_ENC_; e += 256) {
        float acc = 0.0f;
        for (int i = 0; i < NT; ++i)
            acc += ctxp[(size_t)(b * NT + i) * D_ENC_ + e] * scale[i];
        ctx_out[b * D_ENC_ + e] = acc * rz;
    }
    // w[b][t] = exp(s - m) / Z   (masked s = -1e9 -> 0)
    for (int t = tid; t < T_; t += 256)
        w_out[(size_t)b * T_ + t] = expf(scores[(size_t)b * T_ + t] - m) * rz;
}

// ---------------------------------------------------------------------------
extern "C" void kernel_launch(void* const* d_in, const int* in_sizes, int n_in,
                              void* d_out, int out_size, void* d_ws, size_t ws_size,
                              hipStream_t stream) {
    const float* h_enc = (const float*)d_in[0];
    const float* h_dec = (const float*)d_in[1];
    const float* Wq_w  = (const float*)d_in[2];
    const float* Wq_b  = (const float*)d_in[3];
    const float* Wk_w  = (const float*)d_in[4];
    const float* Wk_b  = (const float*)d_in[5];
    const float* v_w   = (const float*)d_in[6];
    const unsigned char* mask = (const unsigned char*)d_in[7];

    char* ws = (char*)d_ws;
    _Float16* Bpack  = (_Float16*)(ws + WS_BPACK);
    float*    qfold  = (float*)(ws + WS_QFOLD);
    float*    scores = (float*)(ws + WS_SCORES);
    float*    ctxp   = (float*)(ws + WS_CTXP);
    float*    mz     = (float*)(ws + WS_MZ);

    float* ctx_out = (float*)d_out;                 // [64,512]
    float* w_out   = (float*)d_out + B_ * D_ENC_;   // [64,4096]

    pack_wk<<<512, 256, 0, stream>>>(Wk_w, Bpack);
    qfold_kernel<<<B_, 256, 0, stream>>>(h_dec, Wq_w, Wq_b, Wk_b, qfold);
    scores_ctx_kernel<<<B_ * NT, 256, SM_BYTES, stream>>>(
        h_enc, Bpack, qfold, v_w, mask, scores, ctxp, mz);
    finalize_kernel<<<B_, 256, 0, stream>>>(scores, ctxp, mz, ctx_out, w_out);
}
